// HyperbolicDistanceLayer_45449343926749
// MI455X (gfx1250) — matrix operand order repaired and checked by reading, not compiled, not run on hardware
//
#include <hip/hip_runtime.h>

typedef _Float16 half16  __attribute__((ext_vector_type(16)));
typedef _Float16 half8   __attribute__((ext_vector_type(8)));
typedef _Float16 half4   __attribute__((ext_vector_type(4)));
typedef float    float8  __attribute__((ext_vector_type(8)));
typedef float    float4v __attribute__((ext_vector_type(4)));

namespace {
constexpr int   kB  = 4;     // batch
constexpr int   kN  = 4096;  // rows (x) == cols (y)
constexpr int   kD  = 128;   // feature dim (K)
constexpr int   BT  = 128;   // output tile edge per workgroup
constexpr int   KC  = 32;    // K chunk staged in LDS (== one WMMA K step)
constexpr int   LDK = KC + 8; // padded LDS row stride in halves (80B: conflict-free)
constexpr float kEps     = 1e-10f;
constexpr float kNormMax = 0.99f;
constexpr float kLn2     = 0.69314718055994530942f;
}

// ---- prep: xs[b,n] = sum_d x^2 ; ys[b,m] = sum_d y^2 (f32, exact path) ----
__global__ __launch_bounds__(256) void hyp_rownorm(const float* __restrict__ x,
                                                   const float* __restrict__ y,
                                                   float* __restrict__ xs,
                                                   float* __restrict__ ys) {
  int row = blockIdx.x * 256 + threadIdx.x;          // 0 .. 2*kB*kN-1
  const float* src = (row < kB * kN) ? (x + (size_t)row * kD)
                                     : (y + (size_t)(row - kB * kN) * kD);
  float acc = 0.0f;
#pragma unroll
  for (int i = 0; i < kD / 4; ++i) {
    float4v v = *(const float4v*)(src + i * 4);
    acc += v[0] * v[0] + v[1] * v[1] + v[2] * v[2] + v[3] * v[3];
  }
  if (row < kB * kN) xs[row] = acc;
  else               ys[row - kB * kN] = acc;
}

// ---- main: 128x128 tile of arctanh-distance per workgroup, split-f16 WMMA ----
__global__ __launch_bounds__(256) void hyp_dist(const float* __restrict__ x,
                                                const float* __restrict__ y,
                                                const float* __restrict__ xs,
                                                const float* __restrict__ ys,
                                                float* __restrict__ out) {
  __shared__ _Float16 Ah[BT * LDK];
  __shared__ _Float16 Al[BT * LDK];
  __shared__ _Float16 Bh[BT * LDK];
  __shared__ _Float16 Bl[BT * LDK];

  const int b      = blockIdx.z;
  const int rowB   = blockIdx.y * BT;   // x-row tile base
  const int colB   = blockIdx.x * BT;   // y-row (output col) tile base
  const int tid    = threadIdx.x;
  const int wave   = tid >> 5;          // 0..7 : 16-row strip per wave
  const int lane   = tid & 31;
  const int lane16 = lane & 15;
  const int laneHi = lane >> 4;         // 0 or 1

  const float* xT = x + ((size_t)b * kN + rowB) * kD;
  const float* yT = y + ((size_t)b * kN + colB) * kD;

  float8 acc[8];
#pragma unroll
  for (int j = 0; j < 8; ++j) {
    float8 z = {0.f, 0.f, 0.f, 0.f, 0.f, 0.f, 0.f, 0.f};
    acc[j] = z;
  }

  for (int kc = 0; kc < kD; kc += KC) {
    // ---- cooperative stage: f32 global -> (hi,lo) f16 split in LDS ----
#pragma unroll
    for (int i = 0; i < (BT * KC / 4) / 256; ++i) {  // 4 iterations
      int idx = tid + i * 256;                       // 0..1023 float4 slots
      int r   = idx >> 3;                            // row 0..127
      int c   = (idx & 7) << 2;                      // col {0,4,...,28}
      float4v vx = *(const float4v*)(xT + (size_t)r * kD + kc + c);
      float4v vy = *(const float4v*)(yT + (size_t)r * kD + kc + c);
      half4 xh, xl, yh, yl;
#pragma unroll
      for (int e = 0; e < 4; ++e) {
        _Float16 h = (_Float16)vx[e];
        xh[e] = h; xl[e] = (_Float16)(vx[e] - (float)h);
        h = (_Float16)vy[e];
        yh[e] = h; yl[e] = (_Float16)(vy[e] - (float)h);
      }
      *(half4*)&Ah[r * LDK + c] = xh;
      *(half4*)&Al[r * LDK + c] = xl;
      *(half4*)&Bh[r * LDK + c] = yh;
      *(half4*)&Bl[r * LDK + c] = yl;
    }
    __syncthreads();

    // ---- A fragments (16x32 f16 layout: lanes0-15 K 0..7/16..23, lanes16-31 +8) ----
    const int arow = wave * 16 + lane16;
    const int ka   = laneHi * 8;
    half8 a0h = *(const half8*)&Ah[arow * LDK + ka];
    half8 a1h = *(const half8*)&Ah[arow * LDK + ka + 16];
    half8 a0l = *(const half8*)&Al[arow * LDK + ka];
    half8 a1l = *(const half8*)&Al[arow * LDK + ka + 16];
    half16 aHi, aLo;
#pragma unroll
    for (int e = 0; e < 8; ++e) {
      aHi[e] = a0h[e]; aHi[e + 8] = a1h[e];
      aLo[e] = a0l[e]; aLo[e + 8] = a1l[e];
    }

#pragma unroll
    for (int j = 0; j < 8; ++j) {
      // B fragment (32x16 f16 layout: N = lane%16, K 0..7/16..23 for lanes0-15, +8 for lanes16-31)
      const int brow = j * 16 + lane16;
      const int kb   = laneHi * 8;
      half8 b0h = *(const half8*)&Bh[brow * LDK + kb];
      half8 b1h = *(const half8*)&Bh[brow * LDK + kb + 16];
      half8 b0l = *(const half8*)&Bl[brow * LDK + kb];
      half8 b1l = *(const half8*)&Bl[brow * LDK + kb + 16];
      half16 bHi, bLo;
#pragma unroll
      for (int e = 0; e < 8; ++e) {
        bHi[e] = b0h[e]; bHi[e + 8] = b1h[e];
        bLo[e] = b0l[e]; bLo[e + 8] = b1l[e];
      }
      // dot ~= xh*yh + xh*yl + xl*yh  (drops only O(2^-22) xl*yl term)
      acc[j] = __builtin_amdgcn_wmma_f32_16x16x32_f16(false, aHi, false, bHi,
                                                      (short)0, acc[j], false, false);
      acc[j] = __builtin_amdgcn_wmma_f32_16x16x32_f16(false, aHi, false, bLo,
                                                      (short)0, acc[j], false, false);
      acc[j] = __builtin_amdgcn_wmma_f32_16x16x32_f16(false, aLo, false, bHi,
                                                      (short)0, acc[j], false, false);
    }
    __syncthreads();
  }

  // ---- epilogue: hyperbolic distance from dot, xs, ys ----
  // C/D layout: acc[j][r] holds D[m = r + 8*(lane/16)][n = lane%16] of tile j.
  const float* xsRow = xs + (size_t)b * kN + rowB + wave * 16;
  const float* ysRow = ys + (size_t)b * kN + colB;
  float xsv[8];
#pragma unroll
  for (int r = 0; r < 8; ++r) xsv[r] = xsRow[r + laneHi * 8];

  float* outT = out + ((size_t)b * kN + rowB + wave * 16) * kN + colB;
#pragma unroll
  for (int j = 0; j < 8; ++j) {
    float ysv = ysRow[j * 16 + lane16];
#pragma unroll
    for (int r = 0; r < 8; ++r) {
      float dot = acc[j][r];
      float xv  = xsv[r];
      float Aq    = 1.0f - 2.0f * dot + ysv;
      float Bq    = 1.0f - xv;
      float denom = 1.0f - 2.0f * dot + xv * ysv;
      float num_sq = Aq * Aq * xv - 2.0f * Aq * Bq * dot + Bq * Bq * ysv;
      num_sq = fmaxf(num_sq, 0.0f);
      float dn   = fmaxf(denom, kEps);
      float diff = __builtin_amdgcn_sqrtf(num_sq) * __builtin_amdgcn_rcpf(dn);
      diff = fminf(fmaxf(diff, kEps), kNormMax);
      // 2*arctanh(d) = ln((1+d)/(1-d)) = v_log_f32(...) * ln2
      float ratio = (1.0f + diff) * __builtin_amdgcn_rcpf(1.0f - diff);
      float res   = __builtin_amdgcn_logf(ratio) * kLn2;
      outT[(size_t)(r + laneHi * 8) * kN + (j * 16 + lane16)] = res;
    }
  }
}

extern "C" void kernel_launch(void* const* d_in, const int* in_sizes, int n_in,
                              void* d_out, int out_size, void* d_ws, size_t ws_size,
                              hipStream_t stream) {
  (void)in_sizes; (void)n_in; (void)out_size; (void)ws_size;
  const float* x = (const float*)d_in[0];
  const float* y = (const float*)d_in[1];
  float* out = (float*)d_out;
  float* xs  = (float*)d_ws;                 // kB*kN floats
  float* ys  = xs + (size_t)kB * kN;         // kB*kN floats (128 KB total ws)

  hyp_rownorm<<<dim3((2 * kB * kN) / 256), dim3(256), 0, stream>>>(x, y, xs, ys);
  hyp_dist<<<dim3(kN / BT, kN / BT, kB), dim3(256), 0, stream>>>(x, y, xs, ys, out);
}
